// P_MLP_21345987461495
// MI455X (gfx1250) — matrix operand order repaired and checked by reading, not compiled, not run on hardware
//
#include <hip/hip_runtime.h>
#include <hip/hip_bf16.h>

typedef __attribute__((ext_vector_type(16))) _Float16 v16h;
typedef __attribute__((ext_vector_type(8)))  _Float16 v8h;
typedef __attribute__((ext_vector_type(8)))  float    v8f;

#define WMMA_F16(a, b, c) \
  __builtin_amdgcn_wmma_f32_16x16x32_f16(false, (a), false, (b), (short)0, (c), false, false)

// Padded LDS strides (halves): 136 -> 272B rows rotate 4 banks/row; 40 -> 80B rotate 20.
#define HS   136   // h1/h2 & W1/W1T row stride
#define HS3  40    // h3 & W2T row stride
// LDS layout sizes (bytes)
#define SZ_C1   (128 * 128 * 4)     // packed per-lane v8f c1
#define SZ_W1   (128 * HS * 2)
#define SZ_W2   (16 * HS * 2)
#define SZ_W2T  (128 * HS3 * 2)
#define SZ_H    (128 * HS * 2)
#define SZ_H3   (128 * HS3 * 2)
#define SMEM_BYTES (SZ_C1 + 2*SZ_W1 + SZ_W2 + SZ_W2T + 4*SZ_H + 2*SZ_H3)  // 309,504 B

__device__ __forceinline__ float fast_tanh(float x) {
#if __has_builtin(__builtin_amdgcn_tanhf)
  return __builtin_amdgcn_tanhf(x);              // native v_tanh_f32 (1 TRANS op)
#elif __has_builtin(__builtin_amdgcn_tanh_f32)
  return __builtin_amdgcn_tanh_f32(x);
#else
  // tanh(x) = 1 - 2/(e^{2x}+1); saturates exactly to +/-1 at the extremes.
  float t = __builtin_amdgcn_exp2f(x * 2.8853900817779268f); // 2*log2(e)
  return 1.0f - 2.0f * __builtin_amdgcn_rcpf(t + 1.0f);
#endif
}

// A operand (16x32 f16): row = r0+(lane&15); lane<16 holds K {k0..+7, k0+16..+23},
// lane>=16 holds K {k0+8..+15, k0+24..+31}. Two aligned 16B LDS reads.
__device__ __forceinline__ v16h lds_loadA(const _Float16* base, int stride,
                                          int r0, int k0, int lane) {
  const _Float16* p = base + (r0 + (lane & 15)) * stride + k0 + ((lane < 16) ? 0 : 8);
  v8h a0 = *(const v8h*)(p);
  v8h a1 = *(const v8h*)(p + 16);
  v16h a;
#pragma unroll
  for (int i = 0; i < 8; ++i) { a[i] = a0[i]; a[i + 8] = a1[i]; }
  return a;
}

// B operand (32x16 f16): column = r0+(lane&15); lane<16 holds K {k0..+7, k0+16..+23},
// lane>=16 holds K {k0+8..+15, k0+24..+31}. Two aligned 16B LDS reads.
__device__ __forceinline__ v16h lds_loadB(const _Float16* base, int stride,
                                          int r0, int k0, int lane) {
  const _Float16* p = base + (r0 + (lane & 15)) * stride + k0 + ((lane < 16) ? 0 : 8);
  v8h b0 = *(const v8h*)(p);
  v8h b1 = *(const v8h*)(p + 16);
  v16h b;
#pragma unroll
  for (int i = 0; i < 8; ++i) { b[i] = b0[i]; b[i + 8] = b1[i]; }
  return b;
}

// 16 contiguous f32 from global (16B aligned) -> f16x16
__device__ __forceinline__ v16h glb_cvt16(const float* p) {
  float t[16];
  *(float4*)(t + 0)  = *(const float4*)(p + 0);
  *(float4*)(t + 4)  = *(const float4*)(p + 4);
  *(float4*)(t + 8)  = *(const float4*)(p + 8);
  *(float4*)(t + 12) = *(const float4*)(p + 12);
  v16h v;
#pragma unroll
  for (int i = 0; i < 16; ++i) v[i] = (_Float16)t[i];
  return v;
}

// Two 8-f32 groups from global (32B apart halves of an A operand) -> f16x16
__device__ __forceinline__ v16h glb_cvt8_8(const float* p0, const float* p1) {
  float t[8], u[8];
  *(float4*)(t + 0) = *(const float4*)(p0 + 0);
  *(float4*)(t + 4) = *(const float4*)(p0 + 4);
  *(float4*)(u + 0) = *(const float4*)(p1 + 0);
  *(float4*)(u + 4) = *(const float4*)(p1 + 4);
  v16h v;
#pragma unroll
  for (int i = 0; i < 8; ++i) { v[i] = (_Float16)t[i]; v[i + 8] = (_Float16)u[i]; }
  return v;
}

__global__ void __launch_bounds__(256, 1)
eqprop_relax_kernel(const float* __restrict__ x,  const float* __restrict__ n1,
                    const float* __restrict__ n2, const float* __restrict__ n3,
                    const float* __restrict__ W0, const float* __restrict__ b0,
                    const float* __restrict__ W1, const float* __restrict__ b1,
                    const float* __restrict__ W2, const float* __restrict__ b2,
                    const int* __restrict__ Tp,
                    float* __restrict__ out1, float* __restrict__ out2,
                    float* __restrict__ out3) {
  extern __shared__ __align__(16) char smem[];
  float*    sC1p = (float*)smem;                       // packed per-lane c1 (v8f)
  _Float16* sW1  = (_Float16*)(smem + SZ_C1);          // [128][HS] W1 row-major
  _Float16* sW1T = sW1  + 128 * HS;                    // [128][HS] W1 transposed
  _Float16* sW2  = sW1T + 128 * HS;                    // [16][HS], rows 10..15 zero
  _Float16* sW2T = sW2  + 16 * HS;                     // [128][HS3], col10 = b1, others>=10 zero
  _Float16* sH1a = sW2T + 128 * HS3;                   // state double buffers
  _Float16* sH1b = sH1a + 128 * HS;
  _Float16* sH2a = sH1b + 128 * HS;
  _Float16* sH2b = sH2a + 128 * HS;
  _Float16* sH3a = sH2b + 128 * HS;                    // [128][HS3]; col10 = 1.0 (bias input)
  _Float16* sH3b = sH3a + 128 * HS3;

  const int tid = threadIdx.x;
  const int m0  = blockIdx.x * 128;

  // ---------------- stage weights / initial state ---------------------------
  for (int i = tid; i < 128 * 128; i += 256) {
    int r = i >> 7, c = i & 127;
    _Float16 v = (_Float16)W1[i];
    sW1[r * HS + c]  = v;
    sW1T[c * HS + r] = v;
    sH1a[r * HS + c] = (_Float16)n1[(size_t)(m0 + r) * 128 + c];
    sH2a[r * HS + c] = (_Float16)n2[(size_t)(m0 + r) * 128 + c];
  }
  for (int i = tid; i < 16 * HS; i += 256) {
    int r = i / HS, c = i % HS;
    sW2[i] = (r < 10 && c < 128) ? (_Float16)W2[r * 128 + c] : (_Float16)0.f;
  }
  for (int i = tid; i < 128 * HS3; i += 256) {
    int n = i / HS3, k = i % HS3;
    _Float16 w = (_Float16)0.f, ha = (_Float16)0.f, hb = (_Float16)0.f;
    if (k < 10) {
      w  = (_Float16)W2[k * 128 + n];
      ha = (_Float16)n3[(size_t)(m0 + n) * 10 + k];
    } else if (k == 10) {          // bias lane: h3[..][10] == 1, W2T[n][10] == b1[n]
      w = (_Float16)b1[n];
      ha = (_Float16)1.f;
      hb = (_Float16)1.f;
    }
    sW2T[i] = w;
    sH3a[i] = ha;
    sH3b[i] = hb;
  }
  __syncthreads();

  const int lane = tid & 31;
  const int wv   = tid >> 5;                 // wave 0..7
  const int rowB = wv * 16;                  // 16 batch rows owned by this wave
  const int cN   = lane & 15;
  const int hio  = (lane < 16) ? 0 : 8;

  // g3 bias (transposed D': element j is output neuron n = j+hio)
  v8f acc3init;
#pragma unroll
  for (int j = 0; j < 8; ++j) {
    int n = j + hio;
    acc3init[j] = (n < 10) ? b2[n] : 0.f;
  }

  // ------------- phase 0: c1^T[n][m] = sum_k W0[n][k] x[m][k] + b0[n] -------
  // A = W0 rows, B = x rows; 24 guard-free K chunks + one explicit tail.
  {
    v8f acc[8];
#pragma unroll
    for (int nt = 0; nt < 8; ++nt)
#pragma unroll
      for (int j = 0; j < 8; ++j) acc[nt][j] = b0[nt * 16 + j + hio];

    const float* xrow = x + (size_t)(m0 + rowB + cN) * 784;
    for (int k0 = 0; k0 < 768; k0 += 32) {
      v16h bx = glb_cvt8_8(xrow + k0 + hio, xrow + k0 + hio + 16);
#pragma unroll
      for (int nt = 0; nt < 8; ++nt) {
        const float* wrow = W0 + (size_t)(nt * 16 + cN) * 784 + k0 + hio;
        acc[nt] = WMMA_F16(glb_cvt8_8(wrow, wrow + 16), bx, acc[nt]);
      }
    }
    {   // tail chunk k0 = 768 (k 768..783 valid, 784..799 are zeros)
      float t[8];
      *(float4*)(t + 0)  = *(const float4*)(xrow + 768 + hio);
      *(float4*)(t + 4)  = *(const float4*)(xrow + 772 + hio);
      v16h bx;
#pragma unroll
      for (int i = 0; i < 16; ++i) bx[i] = (i < 8) ? (_Float16)t[i] : (_Float16)0.f;
#pragma unroll
      for (int nt = 0; nt < 8; ++nt) {
        const float* wrow = W0 + (size_t)(nt * 16 + cN) * 784 + 768 + hio;
        float u[8];
        *(float4*)(u + 0) = *(const float4*)(wrow);
        *(float4*)(u + 4) = *(const float4*)(wrow + 4);
        v16h a;
#pragma unroll
        for (int i = 0; i < 8; ++i) { a[i] = (_Float16)u[i]; a[i + 8] = (_Float16)0.f; }
        acc[nt] = WMMA_F16(a, bx, acc[nt]);
      }
    }
#pragma unroll
    for (int nt = 0; nt < 8; ++nt)
      *(v8f*)(sC1p + ((size_t)(wv * 8 + nt) * 32 + lane) * 8) = acc[nt];
  }
  __syncthreads();

  // ------------- T relaxation steps, all state in LDS ------------------------
  const int T = Tp[0];
  for (int s = 0; s < T; ++s) {
    const bool pp = (s & 1);
    const _Float16* H1r = pp ? sH1b : sH1a;
    const _Float16* H2r = pp ? sH2b : sH2a;
    const _Float16* H3r = pp ? sH3b : sH3a;
    _Float16* H1w = pp ? sH1a : sH1b;
    _Float16* H2w = pp ? sH2a : sH2b;
    _Float16* H3w = pp ? sH3a : sH3b;

    // ---- g1^T = W1^T x h2^T (+c1) and g3^T = W2 x h2^T (+b2): share B ----
    {
      v8f acc[8], acc3 = acc3init;
#pragma unroll
      for (int nt = 0; nt < 8; ++nt)
        acc[nt] = *(const v8f*)(sC1p + ((size_t)(wv * 8 + nt) * 32 + lane) * 8);
#pragma unroll
      for (int kc = 0; kc < 4; ++kc) {
        v16h bh2 = lds_loadB(H2r, HS, rowB, kc * 32, lane);
        acc3 = WMMA_F16(lds_loadA(sW2, HS, 0, kc * 32, lane), bh2, acc3);
#pragma unroll
        for (int nt = 0; nt < 8; ++nt)
          acc[nt] = WMMA_F16(lds_loadA(sW1T, HS, nt * 16, kc * 32, lane), bh2, acc[nt]);
      }
#pragma unroll
      for (int nt = 0; nt < 8; ++nt) {       // one aligned 16B write per tile
        v8h h;
#pragma unroll
        for (int j = 0; j < 8; ++j) h[j] = (_Float16)fast_tanh(acc[nt][j]);
        *(v8h*)(H1w + (rowB + cN) * HS + nt * 16 + hio) = h;
      }
      {
        v8h h;
#pragma unroll
        for (int j = 0; j < 8; ++j) {
          int n = j + hio;
          float t = fast_tanh(acc3[j]);
          h[j] = (n < 10) ? (_Float16)t : ((n == 10) ? (_Float16)1.f : (_Float16)0.f);
        }
        *(v8h*)(H3w + (rowB + cN) * HS3 + hio) = h;
      }
    }

    // ---- g2^T = W1 x h1^T + W2^T x h3^T (bias folded at k=10) ----
    {
      v8f acc[8];
      {
        v16h bh3 = lds_loadB(H3r, HS3, rowB, 0, lane);
        v8f z = {};
#pragma unroll
        for (int nt = 0; nt < 8; ++nt)
          acc[nt] = WMMA_F16(lds_loadA(sW2T, HS3, nt * 16, 0, lane), bh3, z);
      }
#pragma unroll
      for (int kc = 0; kc < 4; ++kc) {
        v16h bh1 = lds_loadB(H1r, HS, rowB, kc * 32, lane);
#pragma unroll
        for (int nt = 0; nt < 8; ++nt)
          acc[nt] = WMMA_F16(lds_loadA(sW1, HS, nt * 16, kc * 32, lane), bh1, acc[nt]);
      }
#pragma unroll
      for (int nt = 0; nt < 8; ++nt) {
        v8h h;
#pragma unroll
        for (int j = 0; j < 8; ++j) h[j] = (_Float16)fast_tanh(acc[nt][j]);
        *(v8h*)(H2w + (rowB + cN) * HS + nt * 16 + hio) = h;
      }
    }
    __syncthreads();
  }

  // ---------------- write outputs (f32) --------------------------------------
  const bool fb = (T & 1);
  const _Float16* F1 = fb ? sH1b : sH1a;
  const _Float16* F2 = fb ? sH2b : sH2a;
  const _Float16* F3 = fb ? sH3b : sH3a;
  for (int i = tid; i < 128 * 128; i += 256) {
    int r = i >> 7, c = i & 127;
    out1[(size_t)(m0 + r) * 128 + c] = (float)F1[r * HS + c];
    out2[(size_t)(m0 + r) * 128 + c] = (float)F2[r * HS + c];
  }
  for (int i = tid; i < 128 * 10; i += 256) {
    int r = i / 10, c = i % 10;
    out3[(size_t)(m0 + r) * 10 + c] = (float)F3[r * HS3 + c];
  }
}

extern "C" void kernel_launch(void* const* d_in, const int* in_sizes, int n_in,
                              void* d_out, int out_size, void* d_ws, size_t ws_size,
                              hipStream_t stream) {
  const float* x  = (const float*)d_in[0];
  // d_in[1] = y (labels, unused)
  const float* n1 = (const float*)d_in[2];
  const float* n2 = (const float*)d_in[3];
  const float* n3 = (const float*)d_in[4];
  const float* W0 = (const float*)d_in[5];
  const float* b0 = (const float*)d_in[6];
  const float* W1 = (const float*)d_in[7];
  const float* b1 = (const float*)d_in[8];
  const float* W2 = (const float*)d_in[9];
  const float* b2 = (const float*)d_in[10];
  const int*   Tp = (const int*)d_in[11];

  const int rowsB  = in_sizes[2] / 128;
  const int blocks = rowsB / 128;

  float* out1 = (float*)d_out;
  float* out2 = out1 + (size_t)rowsB * 128;
  float* out3 = out2 + (size_t)rowsB * 128;

  hipFuncSetAttribute(reinterpret_cast<const void*>(eqprop_relax_kernel),
                      hipFuncAttributeMaxDynamicSharedMemorySize, SMEM_BYTES);
  eqprop_relax_kernel<<<dim3(blocks), dim3(256), SMEM_BYTES, stream>>>(
      x, n1, n2, n3, W0, b0, W1, b1, W2, b2, Tp, out1, out2, out3);
}
